// SO2ConvolutionMoE_53051436040415
// MI455X (gfx1250) — compile-verified, matrix operand order checked by reading, not run on hardware
//
#include <hip/hip_runtime.h>
#include <hip/hip_bf16.h>

typedef __attribute__((ext_vector_type(16))) _Float16 v16h;
typedef __attribute__((ext_vector_type(8)))  _Float16 v8h;
typedef __attribute__((ext_vector_type(4)))  _Float16 v4h;
typedef __attribute__((ext_vector_type(8)))  float    v8f;
typedef unsigned int u32x4 __attribute__((ext_vector_type(4)));
typedef int          i32x4 __attribute__((ext_vector_type(4)));
typedef int          i32x8 __attribute__((ext_vector_type(8)));

#define N_EDGES   12288
#define N_SYS     16
#define N_EXP     8
#define ROW_W     2432          // 19 * 128 elements per edge row (in and out)

// ---------------------------------------------------------------------------
// Bucketing: counting sort of edges by system (16 bins)
// ---------------------------------------------------------------------------
__global__ __launch_bounds__(32) void so2_init(int* counts, int* cursor) {
    if (threadIdx.x < N_SYS) { counts[threadIdx.x] = 0; cursor[threadIdx.x] = 0; }
}

__global__ __launch_bounds__(256) void so2_hist(const int* __restrict__ eb, int* counts) {
    int i = blockIdx.x * 256 + threadIdx.x;
    if (i < N_EDGES) atomicAdd(&counts[eb[i]], 1);
}

__global__ __launch_bounds__(32) void so2_scan(const int* __restrict__ counts, int* sysoff) {
    if (threadIdx.x == 0) {
        int a = 0;
        for (int s = 0; s < N_SYS; ++s) { sysoff[s] = a; a += counts[s]; }
        sysoff[N_SYS] = a;
    }
}

__global__ __launch_bounds__(256) void so2_scatter(const int* __restrict__ eb,
                                                   const int* __restrict__ sysoff,
                                                   int* cursor, int* perm) {
    int i = blockIdx.x * 256 + threadIdx.x;
    if (i < N_EDGES) {
        int b = eb[i];
        int p = atomicAdd(&cursor[b], 1);
        perm[sysoff[b] + p] = i;
    }
}

// ---------------------------------------------------------------------------
// Convert x to f16 (single pass, float4 -> 4 halves)
// ---------------------------------------------------------------------------
__global__ __launch_bounds__(256) void so2_cvt(const float* __restrict__ x,
                                               _Float16* __restrict__ xh, int n) {
    int idx = (blockIdx.x * 256 + threadIdx.x) * 4;
    if (idx >= n) return;
    float4 f = *(const float4*)(x + idx);
    v4h h;
    h[0] = (_Float16)f.x; h[1] = (_Float16)f.y;
    h[2] = (_Float16)f.z; h[3] = (_Float16)f.w;
    *(v4h*)(xh + idx) = h;
}

// ---------------------------------------------------------------------------
// Per-system weight mixing + complex rearrangement + transpose, f32 -> f16.
// Source W: [8][SR][D].  Target Wt: [16][D (out)][D (in)] where
//   W'[i][o] = W[i][o]            (i <  H)
//   W'[i][o] = -W[i-H][o+H]       (i >= H, o <  H)
//   W'[i][o] =  W[i-H][o-H]       (i >= H, o >= H)
// For the m=0 branch pass H = D (identity). 32 | H always.
// ---------------------------------------------------------------------------
template<int D, int H, int SR>
__global__ __launch_bounds__(256) void so2_mix(const float* __restrict__ W,
                                               const float* __restrict__ ce,
                                               _Float16* __restrict__ Wt) {
    __shared__ float tile[N_EXP][32][33];
    __shared__ float ceL[N_SYS * N_EXP];
    const int tid = threadIdx.x;
    const int i0 = blockIdx.x * 32;      // target input-dim base
    const int o0 = blockIdx.y * 32;      // target output-dim base

    int si0, so0; float sgn;
    if (i0 < H) { si0 = i0; so0 = o0; sgn = 1.0f; }
    else {
        si0 = i0 - H;
        if (o0 < H) { so0 = o0 + H; sgn = -1.0f; }
        else        { so0 = o0 - H; sgn =  1.0f; }
    }

    if (tid < N_SYS * N_EXP) ceL[tid] = ce[tid];
    #pragma unroll
    for (int k = 0; k < N_EXP; ++k) {
        #pragma unroll
        for (int rep = 0; rep < 4; ++rep) {
            int idx = rep * 256 + tid;
            int i = idx >> 5, o = idx & 31;
            tile[k][i][o] = W[((size_t)k * SR + si0 + i) * D + so0 + o];
        }
    }
    __syncthreads();

    for (int s = 0; s < N_SYS; ++s) {
        #pragma unroll
        for (int rep = 0; rep < 4; ++rep) {
            int idx = rep * 256 + tid;
            int o = idx >> 5, i = idx & 31;     // i fastest -> coalesced stores
            float v = 0.0f;
            #pragma unroll
            for (int k = 0; k < N_EXP; ++k) v += ceL[s * N_EXP + k] * tile[k][i][o];
            Wt[((size_t)s * D + (o0 + o)) * D + (i0 + i)] = (_Float16)(sgn * v);
        }
    }
}

// Mixed bias for the m=0 branch: bias[s][o] = sum_k ce[s][k] * b[k][o]
__global__ __launch_bounds__(256) void so2_bias(const float* __restrict__ b,
                                                const float* __restrict__ ce,
                                                float* __restrict__ bias) {
    int idx = blockIdx.x * 256 + threadIdx.x;
    if (idx >= N_SYS * 640) return;
    int s = idx / 640, o = idx % 640;
    float v = 0.0f;
    #pragma unroll
    for (int k = 0; k < N_EXP; ++k) v += ce[s * N_EXP + k] * b[k * 640 + o];
    bias[idx] = v;
}

// ---------------------------------------------------------------------------
// WMMA fragment load: row-major source, one 16-wide row per lane.
// Lane L (L<16) holds K = [0..7] and [16..23]; lane L>=16 holds K = [8..15]
// and [24..31] of row L%16 — per the CDNA5 16-bit A/B VGPR layout.
// Works for global or LDS pointers (addrspace inferred -> ds_load_b128).
// ---------------------------------------------------------------------------
__device__ __forceinline__ v16h load_frag16(const _Float16* p, int hs) {
    v8h lo = *(const v8h*)(p + hs * 8);
    v8h hi = *(const v8h*)(p + 16 + hs * 8);
    v16h r;
    #pragma unroll
    for (int i = 0; i < 8; ++i) { r[i] = lo[i]; r[8 + i] = hi[i]; }
    return r;
}

// ---------------------------------------------------------------------------
// TDM: stage a 128(rows) x 128(K halves) tile of Wt into LDS.
// D# per CDNA5 ISA:
//   group0: count=1 | lds_addr | global_addr(57b) | type=2
//   group1: data_size=1(2B), tensor_dim0=IN, tensor_dim1=128,
//           tile_dim0=128, tile_dim1=128, tensor_dim0_stride=IN
//   groups 2/3: zero (2D tensor; tile_dim2=0 -> unused)
// ---------------------------------------------------------------------------
__device__ __forceinline__ void tdm_load_tile(const _Float16* gptr, unsigned lds_off,
                                              int in_elems) {
    unsigned long long ga = (unsigned long long)(uintptr_t)gptr;
    u32x4 g0;
    g0[0] = 1u;                                         // count=1, user descriptor
    g0[1] = lds_off;                                    // LDS byte address
    g0[2] = (unsigned)(ga & 0xFFFFFFFFu);               // global_addr[31:0]
    g0[3] = (unsigned)((ga >> 32) & 0x01FFFFFFu) | (2u << 30);  // addr[56:32]|type=2
    i32x8 g1;
    g1[0] = (int)(1u << 16);                            // data_size=1 (2 bytes)
    g1[1] = (int)((unsigned)in_elems << 16);            // tensor_dim0[15:0]
    g1[2] = (int)(128u << 16);                          // tensor_dim1[15:0]=128
    g1[3] = (int)(128u << 16);                          // tile_dim0=128
    g1[4] = 128;                                        // tile_dim1=128, tile_dim2=0
    g1[5] = in_elems;                                   // tensor_dim0_stride[31:0]
    g1[6] = 0;
    g1[7] = 0;
    i32x4 gz = {0, 0, 0, 0};
#if defined(__clang_major__) && (__clang_major__ >= 23)
    i32x8 gz8 = {0, 0, 0, 0, 0, 0, 0, 0};
    __builtin_amdgcn_tensor_load_to_lds(g0, g1, gz, gz, gz8, 0);
#else
    __builtin_amdgcn_tensor_load_to_lds(g0, g1, gz, gz, 0);
#endif
}

// ---------------------------------------------------------------------------
// Grouped GEMM over systems: out[perm[rows], OOFF:OOFF+OUT] =
//   xh[perm[rows], XOFF:XOFF+IN] @ Wt[s]^T  (+ bias)
// Block: 256 thr = 8 waves arranged 4(M) x 2(N); block tile 128(M) x 128(N).
// Wave tile: 32(M) x 64(N) = 8 accumulators.  B tile is staged to LDS by the
// Tensor Data Mover (double-buffered, TENSORcnt + barrier), A rows gathered
// directly from global through perm.
// ---------------------------------------------------------------------------
template<int IN, int OUT, int XOFF, int OOFF, bool BIAS>
__global__ __launch_bounds__(256) void so2_gemm(const _Float16* __restrict__ xh,
                                                const _Float16* __restrict__ Wt,
                                                const float* __restrict__ bias,
                                                const int* __restrict__ perm,
                                                const int* __restrict__ sysoff,
                                                float* __restrict__ out) {
    constexpr int KC = 128;              // K chunk (halves) staged per TDM op
    constexpr int NC = IN / KC;          // chunks: 5 / 8 / 6
    __shared__ _Float16 bt[2][128 * KC]; // 2 x 32 KB double buffer

    const int tid  = threadIdx.x;
    const int lane = tid & 31;
    const int wid  = tid >> 5;
    const int wm   = wid >> 1;           // wave row (0..3)
    const int wn   = wid & 1;            // wave col (0..1)
    const int hs   = (lane >> 4) & 1;
    const int r    = lane & 15;

    // map blockIdx.x -> (system, local M tile of 128 rows)
    int t = blockIdx.x;
    int s = 0, base = 0, cnt = 0;
    for (; s < N_SYS; ++s) {
        base = sysoff[s];
        cnt  = sysoff[s + 1] - base;
        int nt = (cnt + 127) >> 7;
        if (t < nt) break;
        t -= nt;
    }
    if (s >= N_SYS) return;
    const int m_base  = base + t * 128;
    const int sys_end = base + cnt;

    const int ncb = blockIdx.y * 128;                      // block's N base
    const _Float16* Wblk = Wt + (size_t)s * OUT * IN + (size_t)ncb * IN;

    // per-lane A rows (two 16-row fragments, clamped gather through perm)
    int aslot0 = m_base + wm * 32 + r;
    int aslot1 = aslot0 + 16;
    if (aslot0 >= sys_end) aslot0 = sys_end - 1;
    if (aslot1 >= sys_end) aslot1 = sys_end - 1;
    const _Float16* ap0 = xh + (size_t)perm[aslot0] * ROW_W + XOFF;
    const _Float16* ap1 = xh + (size_t)perm[aslot1] * ROW_W + XOFF;

    v8f acc[2][4];
    #pragma unroll
    for (int i = 0; i < 2; ++i)
        #pragma unroll
        for (int j = 0; j < 4; ++j) acc[i][j] = (v8f){};

    if (wid == 0)
        tdm_load_tile(Wblk, (unsigned)(uintptr_t)&bt[0][0], IN);

    for (int c = 0; c < NC; ++c) {
        if (wid == 0) {
            if (c + 1 < NC) {
                tdm_load_tile(Wblk + (c + 1) * KC,
                              (unsigned)(uintptr_t)&bt[(c + 1) & 1][0], IN);
                __builtin_amdgcn_s_wait_tensorcnt(1);   // chunk c complete
            } else {
                __builtin_amdgcn_s_wait_tensorcnt(0);
            }
        }
        __syncthreads();                                 // publish buffer c&1

        const _Float16* bb = &bt[c & 1][0] + (wn * 64 + r) * KC;
        const _Float16* a0p = ap0 + c * KC;
        const _Float16* a1p = ap1 + c * KC;
        #pragma unroll
        for (int kk = 0; kk < KC; kk += 32) {
            v16h a0 = load_frag16(a0p + kk, hs);
            v16h a1 = load_frag16(a1p + kk, hs);
            v16h b0 = load_frag16(bb + kk,           hs);
            v16h b1 = load_frag16(bb + kk + 16 * KC, hs);
            v16h b2 = load_frag16(bb + kk + 32 * KC, hs);
            v16h b3 = load_frag16(bb + kk + 48 * KC, hs);
            acc[0][0] = __builtin_amdgcn_wmma_f32_16x16x32_f16(false, a0, false, b0, (short)0, acc[0][0], false, false);
            acc[0][1] = __builtin_amdgcn_wmma_f32_16x16x32_f16(false, a0, false, b1, (short)0, acc[0][1], false, false);
            acc[0][2] = __builtin_amdgcn_wmma_f32_16x16x32_f16(false, a0, false, b2, (short)0, acc[0][2], false, false);
            acc[0][3] = __builtin_amdgcn_wmma_f32_16x16x32_f16(false, a0, false, b3, (short)0, acc[0][3], false, false);
            acc[1][0] = __builtin_amdgcn_wmma_f32_16x16x32_f16(false, a1, false, b0, (short)0, acc[1][0], false, false);
            acc[1][1] = __builtin_amdgcn_wmma_f32_16x16x32_f16(false, a1, false, b1, (short)0, acc[1][1], false, false);
            acc[1][2] = __builtin_amdgcn_wmma_f32_16x16x32_f16(false, a1, false, b2, (short)0, acc[1][2], false, false);
            acc[1][3] = __builtin_amdgcn_wmma_f32_16x16x32_f16(false, a1, false, b3, (short)0, acc[1][3], false, false);
        }
        __syncthreads();                                 // done reading buffer c&1
    }

    float bj[4] = {0.0f, 0.0f, 0.0f, 0.0f};
    if constexpr (BIAS) {
        #pragma unroll
        for (int j = 0; j < 4; ++j) bj[j] = bias[s * 640 + ncb + wn * 64 + j * 16 + r];
    }

    #pragma unroll
    for (int i = 0; i < 2; ++i) {
        #pragma unroll
        for (int v = 0; v < 8; ++v) {
            int slot = m_base + wm * 32 + i * 16 + hs * 8 + v;  // C: M = v + 8*hs
            if (slot < sys_end) {
                int e = perm[slot];
                float* orow = out + (size_t)e * ROW_W + OOFF + ncb + wn * 64;
                #pragma unroll
                for (int j = 0; j < 4; ++j)
                    orow[j * 16 + r] = acc[i][j][v] + bj[j];
            }
        }
    }
}

// ---------------------------------------------------------------------------
extern "C" void kernel_launch(void* const* d_in, const int* in_sizes, int n_in,
                              void* d_out, int out_size, void* d_ws, size_t ws_size,
                              hipStream_t stream) {
    const float* x  = (const float*)d_in[0];            // [12288][19][128]
    const float* ce = (const float*)d_in[2];            // [16][8]
    const int*   eb = (const int*)  d_in[3];            // [12288]
    const float* W0 = (const float*)d_in[4];            // [8][640][640]
    const float* b0 = (const float*)d_in[5];            // [8][640]
    const float* W1 = (const float*)d_in[6];            // [8][512][1024]
    const float* W2 = (const float*)d_in[7];            // [8][384][768]
    float* out = (float*)d_out;

    char* ws = (char*)d_ws;
    int*       counts = (int*)(ws + 0);
    int*       cursor = (int*)(ws + 64);
    int*       sysoff = (int*)(ws + 128);
    int*       perm   = (int*)(ws + 512);
    float*     bias   = (float*)(ws + 49664);
    _Float16*  xh     = (_Float16*)(ws + 90624);
    _Float16*  w0t    = (_Float16*)(ws + 59859456ull);
    _Float16*  w1t    = (_Float16*)(ws + 72966656ull);
    _Float16*  w2t    = (_Float16*)(ws + 106521088ull);

    so2_init<<<1, 32, 0, stream>>>(counts, cursor);
    so2_hist<<<48, 256, 0, stream>>>(eb, counts);
    so2_scan<<<1, 32, 0, stream>>>(counts, sysoff);
    so2_scatter<<<48, 256, 0, stream>>>(eb, sysoff, cursor, perm);

    so2_cvt<<<(N_EDGES * ROW_W) / 4 / 256, 256, 0, stream>>>(x, xh, N_EDGES * ROW_W);

    so2_mix< 640, 640, 640><<<dim3(20, 20), 256, 0, stream>>>(W0, ce, w0t);
    so2_mix<1024, 512, 512><<<dim3(32, 32), 256, 0, stream>>>(W1, ce, w1t);
    so2_mix< 768, 384, 384><<<dim3(24, 24), 256, 0, stream>>>(W2, ce, w2t);
    so2_bias<<<(N_SYS * 640 + 255) / 256, 256, 0, stream>>>(b0, ce, bias);

    // max M tiles of 128 rows = 12288/128 + 15 slack -> 112
    so2_gemm< 640,  640,    0,    0, true ><<<dim3(112, 5), 256, 0, stream>>>(xh, w0t, bias, perm, sysoff, out);
    so2_gemm<1024, 1024,  640,  640, false><<<dim3(112, 8), 256, 0, stream>>>(xh, w1t, nullptr, perm, sysoff, out);
    so2_gemm< 768,  768, 1664, 1664, false><<<dim3(112, 6), 256, 0, stream>>>(xh, w2t, nullptr, perm, sysoff, out);
}
